// VideoSelfAttention_84731114815659
// MI455X (gfx1250) — compile-verified, matrix operand order checked by reading, not run on hardware
//
#include <hip/hip_runtime.h>
#include <hip/hip_bf16.h>
#include <math.h>

// ---------------------------------------------------------------------------
// VideoSelfAttention for MI455X (gfx1250), wave32 + WMMA f16 (f32 accum),
// async LDS staging via global_load_async_to_lds_b128 (ASYNCcnt).
//   x:(32,768,16,16) f32, w_qkv:(2304,768) f32, w_proj:(768,768) f32
//   batch=2 videos, 16 frames, 256 tokens/frame, 12 heads, head dim 64.
// Workspace (256-aligned, ~67.6 MB):
//   wq_h 3.54MB | wp_h 1.18MB | xh 12.6MB | q 12.6MB | k 12.6MB | vT 12.6MB
//   | xo 12.6MB
// Fragment packing: 16-bit WMMA A/B lanes want element runs {0..7,16..23}
// (lane<16) / {8..15,24..31} (lane>=16) of each 32-wide K chunk. Staged data
// is stored with 8-element runs 1<->2 of each 32-chunk swapped so every
// fragment is one contiguous 32-byte LDS/global load.
// ---------------------------------------------------------------------------

typedef __attribute__((ext_vector_type(16))) _Float16 v16h;
typedef __attribute__((ext_vector_type(8)))  _Float16 v8h;
typedef __attribute__((ext_vector_type(8)))  float    v8f;

#define WMMA_F16(A, B, C) \
  __builtin_amdgcn_wmma_f32_16x16x32_f16(false, (A), false, (B), (short)0, (C), false, false)

static __device__ __forceinline__ v16h cat8(v8h lo, v8h hi) {
  return __builtin_shufflevector(lo, hi, 0,1,2,3,4,5,6,7,8,9,10,11,12,13,14,15);
}
// Fragment from plain row-major f16 (two 16B runs, 16 ch apart).
static __device__ __forceinline__ v16h load_frag(const _Float16* p) {
  return cat8(*(const v8h*)p, *(const v8h*)(p + 16));
}
// run-swap within a 32-chunk: 8-element runs 0,1,2,3 -> 0,2,1,3
static __device__ __forceinline__ int runswap(int g) {
  return ((g & 1) << 1) | (g >> 1);
}
// LDS byte offset of a __shared__ object (flat shared addr low 32 bits).
static __device__ __forceinline__ unsigned lds_off(const void* p) {
  return (unsigned)(uintptr_t)p;
}
// Async 16B copy global -> LDS, per lane. Tracked by ASYNCcnt.
static __device__ __forceinline__ void async_cp16(unsigned lds, const void* g) {
  asm volatile("global_load_async_to_lds_b128 %0, %1, off"
               :: "v"(lds), "v"(g) : "memory");
}
static __device__ __forceinline__ void async_wait0() {
  asm volatile("s_wait_asynccnt 0x0" ::: "memory");
}

// ---------------------------------------------------------------------------
// K0: _mp_weight — per-row normalize, emit f16. One wave per 768-wide row.
// ---------------------------------------------------------------------------
__global__ __launch_bounds__(256) void k_prep_w(const float* __restrict__ w,
                                                _Float16* __restrict__ o, int rows) {
  const int row  = (int)(blockIdx.x * blockDim.x + threadIdx.x) >> 5;
  const int lane = threadIdx.x & 31;
  if (row >= rows) return;
  const float* wr = w + (size_t)row * 768;
  float ss = 0.f;
  for (int j = lane; j < 768; j += 32) { float t = wr[j]; ss += t * t; }
  #pragma unroll
  for (int m = 16; m >= 1; m >>= 1) ss += __shfl_xor(ss, m, 32);
  const float scale =
      1.0f / ((1e-4f + sqrtf(ss) * 0.036084391824351615f) * 27.712812921102035f);
  _Float16* orow = o + (size_t)row * 768;
  for (int j = lane; j < 768; j += 32) orow[j] = (_Float16)(wr[j] * scale);
}

// ---------------------------------------------------------------------------
// K0b: transpose x (b,ch,p) f32 -> xh (token, ch) f16 with A-fragment
// permutation inside each 32-channel chunk. grid = 32 images * 24 chunks.
// ---------------------------------------------------------------------------
__global__ __launch_bounds__(256) void k_xpose(const float* __restrict__ x,
                                               _Float16* __restrict__ xh) {
  const int b  = blockIdx.x / 24;
  const int kc = blockIdx.x % 24;
  const int t  = threadIdx.x;
  const int cl = t >> 3;                      // channel in chunk 0..31
  const int pg = (t & 7) * 32;                // token base within image
  const int cp = ((cl >> 3) & 1) * 16 + (cl & 7) + ((cl >> 4) << 3);
  const float* xp = x + ((size_t)b * 768 + kc * 32 + cl) * 256 + pg;
  _Float16* op = xh + (size_t)(b * 256 + pg) * 768 + kc * 32 + cp;
  #pragma unroll
  for (int p = 0; p < 32; ++p)
    op[(size_t)p * 768] = (_Float16)xp[p];
}

// ---------------------------------------------------------------------------
// K1: QKV GEMM + per-head L2 normalize + RoPE (q,k) + scatter to q/k/vT.
// Wave tile: 16 tokens x 64 out-channels (one (qkv,head) chunk). Weights are
// workgroup-uniform -> double-buffered async LDS staging shared by 8 waves.
// grid = (64 token tiles of 128, 36 chunks), block = 256 (8 waves).
// ---------------------------------------------------------------------------
__global__ __launch_bounds__(256) void k_qkv(const _Float16* __restrict__ xh,
                                             const _Float16* __restrict__ wq,
                                             _Float16* __restrict__ qb,
                                             _Float16* __restrict__ kb,
                                             _Float16* __restrict__ vt) {
  __shared__ __attribute__((aligned(64))) _Float16 Wsh[2][64][32];
  const int lane = threadIdx.x & 31;
  const int wave = threadIdx.x >> 5;
  const int ln = lane & 15, lh = lane >> 4;
  const int t0 = blockIdx.x * 128 + wave * 16;     // token tile (within 1 image)
  const int chunk = blockIdx.y;                    // 0..35 = sel*12 + head
  const int sel = chunk / 12, head = chunk % 12;
  const int obase = chunk * 64;

  const int st = threadIdx.x;
  const int srow = st >> 2, sg = st & 3, sgp = runswap(sg);
  const _Float16* wrow = wq + (size_t)(obase + srow) * 768 + sg * 8;
  const unsigned wdst[2] = { lds_off(&Wsh[0][srow][sgp * 8]),
                             lds_off(&Wsh[1][srow][sgp * 8]) };

  // prologue: stage weight chunk 0 (async DMA to LDS)
  async_cp16(wdst[0], wrow);
  async_wait0();
  __syncthreads();

  const _Float16* arow = xh + (size_t)(t0 + ln) * 768 + (lh << 4);
  v8f acc[4] = {};
  for (int kc = 0; kc < 24; ++kc) {
    const int buf = kc & 1;
    if (kc + 1 < 24)
      async_cp16(wdst[buf ^ 1], wrow + (kc + 1) * 32);
    if (kc + 2 < 24)
      __builtin_prefetch(wrow + (kc + 2) * 32, 0, 2);
    const v16h a = *(const v16h*)(arow + kc * 32);     // pre-permuted A frag
    #pragma unroll
    for (int nt = 0; nt < 4; ++nt) {
      const v16h bf = *(const v16h*)&Wsh[buf][nt * 16 + ln][lh << 4];
      acc[nt] = WMMA_F16(a, bf, acc[nt]);
    }
    async_wait0();
    __syncthreads();
  }

  // ---- normalize each token row over its 64 head channels ----
  float scl[8];
  #pragma unroll
  for (int vv = 0; vv < 8; ++vv) {
    float ssum = 0.f;
    #pragma unroll
    for (int j = 0; j < 4; ++j) { float t = acc[j][vv]; ssum += t * t; }
    #pragma unroll
    for (int m2 = 8; m2 >= 1; m2 >>= 1) ssum += __shfl_xor(ssum, m2, 16);
    scl[vv] = 1.0f / (1e-4f + sqrtf(ssum) * 0.125f);   // sqrt(c)=8
  }
  #pragma unroll
  for (int j = 0; j < 4; ++j)
    #pragma unroll
    for (int vv = 0; vv < 8; ++vv) acc[j][vv] *= scl[vv];

  // ---- RoPE for q,k: angle = frame * theta^(-d/32), pairs (d, d+32) ----
  if (sel < 2) {
    const float frame = (float)((t0 >> 8) & 15);
    #pragma unroll
    for (int j = 0; j < 2; ++j) {
      const float d = (float)(16 * j + ln);
      const float ang = frame * __expf(d * -0.28782313662425572f); // ln(1e4)/32
      const float cs = __cosf(ang), sn = __sinf(ang);
      #pragma unroll
      for (int vv = 0; vv < 8; ++vv) {
        const float x1 = acc[j][vv], x2 = acc[j + 2][vv];
        acc[j][vv]     = x1 * cs - x2 * sn;
        acc[j + 2][vv] = x2 * cs + x1 * sn;
      }
    }
  }
  const float qs = (sel == 0) ? 0.125f : 1.0f;     // fold 1/sqrt(c) into q
  const int bm = (t0 >> 12) * 12 + head;           // video * 12 + head
  #pragma unroll
  for (int j = 0; j < 4; ++j)
    #pragma unroll
    for (int vv = 0; vv < 8; ++vv) {
      const int tokv = (t0 + vv + 8 * lh) & 4095;  // token within video
      const int ch = 16 * j + ln;
      const _Float16 val = (_Float16)(acc[j][vv] * qs);
      if (sel == 0)      qb[((size_t)bm * 4096 + tokv) * 64 + ch] = val;
      else if (sel == 1) kb[((size_t)bm * 4096 + tokv) * 64 + ch] = val;
      else               vt[((size_t)bm * 64 + ch) * 4096 + tokv] = val; // V^T
    }
}

// ---------------------------------------------------------------------------
// K2: flash attention, frame-block-causal (256-token blocks fully valid -> no
// element masking). Wave tile: 16 queries; KV streamed 32 keys/iter through
// double-buffered async-staged LDS shared by all 8 waves (same frame range).
// grid = (32 q-tiles of 128, 24 bm), block = 256 (8 waves).
// ---------------------------------------------------------------------------
__global__ __launch_bounds__(256) void k_attn(const _Float16* __restrict__ qb,
                                              const _Float16* __restrict__ kb,
                                              const _Float16* __restrict__ vt,
                                              _Float16* __restrict__ xo) {
  __shared__ __attribute__((aligned(64))) _Float16 Ksh[2][32][64];
  __shared__ __attribute__((aligned(64))) _Float16 Vsh[2][64][32];
  __shared__ __attribute__((aligned(64))) _Float16 Psh[8][16][32]; // per-wave
  const int lane = threadIdx.x & 31;
  const int wave = threadIdx.x >> 5;
  const int ln = lane & 15, lh = lane >> 4;
  const int bm = blockIdx.y;
  const int q0 = blockIdx.x * 128 + wave * 16;
  const _Float16* q = qb + (size_t)bm * 4096 * 64;
  const _Float16* k = kb + (size_t)bm * 4096 * 64;
  const _Float16* v = vt + (size_t)bm * 64 * 4096;

  // cooperative staging addresses (per thread)
  const int st = threadIdx.x;
  const int kkey = st >> 3, kgg = st & 7;          // K: key row, 8-half group
  const int kg = kgg & 3, kgp = runswap(kg);
  const int vch = st >> 2, vkg = st & 3;           // V: channel row, key group
  const int vkp = runswap(vkg);
  const _Float16* krow = k + (size_t)kkey * 64 + kgg * 8;
  const _Float16* vrow = v + (size_t)vch * 4096 + vkg * 8;
  const unsigned kdst[2] = { lds_off(&Ksh[0][kkey][(kgg >> 2) * 32 + kgp * 8]),
                             lds_off(&Ksh[1][kkey][(kgg >> 2) * 32 + kgp * 8]) };
  const unsigned vdst[2] = { lds_off(&Vsh[0][vch][vkp * 8]),
                             lds_off(&Vsh[1][vch][vkp * 8]) };

  v16h aq[2];
  #pragma unroll
  for (int cc = 0; cc < 2; ++cc)
    aq[cc] = load_frag(q + (size_t)(q0 + ln) * 64 + cc * 32 + (lh << 3));

  float mrow[8], lrow[8];
  #pragma unroll
  for (int vv = 0; vv < 8; ++vv) { mrow[vv] = -1e30f; lrow[vv] = 0.f; }
  v8f o[4] = {};

  const int nch = ((q0 >> 8) + 1) * 8;             // (frame+1)*256/32 chunks

  // prologue: stage chunk 0 (async DMA to LDS)
  async_cp16(kdst[0], krow);
  async_cp16(vdst[0], vrow);
  async_wait0();
  __syncthreads();

  for (int c = 0; c < nch; ++c) {
    const int buf = c & 1;
    if (c + 1 < nch) {   // async-stage next chunk into the other buffer
      async_cp16(kdst[buf ^ 1], krow + (size_t)(c + 1) * 32 * 64);
      async_cp16(vdst[buf ^ 1], vrow + (c + 1) * 32);
    }
    if (c + 2 < nch) {   // latency cover (global_prefetch_b8)
      __builtin_prefetch(krow + (size_t)(c + 2) * 32 * 64, 0, 2);
      __builtin_prefetch(vrow + (c + 2) * 32, 0, 2);
    }
    v8f s[2] = {};
    #pragma unroll
    for (int kt = 0; kt < 2; ++kt)
      #pragma unroll
      for (int cc = 0; cc < 2; ++cc) {             // K dim 64 = 2 x 32
        const v16h bk = *(const v16h*)&Ksh[buf][kt * 16 + ln][cc * 32 + (lh << 4)];
        s[kt] = WMMA_F16(aq[cc], bk, s[kt]);
      }
    // ---- online softmax (rows live in 16-lane halves of C layout) ----
    float fac[8];
    #pragma unroll
    for (int vv = 0; vv < 8; ++vv) {
      float mx = fmaxf(s[0][vv], s[1][vv]);
      #pragma unroll
      for (int m2 = 8; m2 >= 1; m2 >>= 1) mx = fmaxf(mx, __shfl_xor(mx, m2, 16));
      const float mnew = fmaxf(mrow[vv], mx);
      fac[vv] = __expf(mrow[vv] - mnew);
      mrow[vv] = mnew;
      const float p0 = __expf(s[0][vv] - mnew);
      const float p1 = __expf(s[1][vv] - mnew);
      s[0][vv] = p0; s[1][vv] = p1;
      float rs = p0 + p1;
      #pragma unroll
      for (int m2 = 8; m2 >= 1; m2 >>= 1) rs += __shfl_xor(rs, m2, 16);
      lrow[vv] = lrow[vv] * fac[vv] + rs;
    }
    #pragma unroll
    for (int j = 0; j < 4; ++j)
      #pragma unroll
      for (int vv = 0; vv < 8; ++vv) o[j][vv] *= fac[vv];
    // ---- C layout -> A layout via per-wave LDS (same-wave DS is in-order) --
    #pragma unroll
    for (int kt = 0; kt < 2; ++kt) {
      const int N = kt * 16 + ln;
      const int cp = ((N >> 3) & 1) * 16 + (N & 7) + ((N >> 4) << 3);
      #pragma unroll
      for (int vv = 0; vv < 8; ++vv)
        Psh[wave][vv + 8 * lh][cp] = (_Float16)s[kt][vv];
    }
    const v16h ap = *(const v16h*)&Psh[wave][ln][lh << 4];
    #pragma unroll
    for (int nc = 0; nc < 4; ++nc) {               // O(16x64) += P(16x32)*V
      const v16h bvf = *(const v16h*)&Vsh[buf][nc * 16 + ln][lh << 4];
      o[nc] = WMMA_F16(ap, bvf, o[nc]);
    }
    async_wait0();   // wave's staging DMA complete before the barrier
    __syncthreads();
  }
  // ---- epilogue: 1/l, scatter to (token, c_idx*12+head) f16 for proj ----
  const int bvid = bm / 12, head = bm % 12;
  #pragma unroll
  for (int j = 0; j < 4; ++j)
    #pragma unroll
    for (int vv = 0; vv < 8; ++vv) {
      const int tokv = q0 + vv + 8 * lh;
      const int cidx = 16 * j + ln;
      xo[((size_t)(bvid * 4096 + tokv)) * 768 + cidx * 12 + head] =
          (_Float16)(o[j][vv] / lrow[vv]);
    }
}

// ---------------------------------------------------------------------------
// K3: output projection GEMM + residual mix, fp32 out. Weights async-staged.
// grid = (64 token tiles of 128, 12 out-chunks of 64), block = 256.
// ---------------------------------------------------------------------------
__global__ __launch_bounds__(256) void k_proj(const _Float16* __restrict__ xo,
                                              const _Float16* __restrict__ wp,
                                              const float* __restrict__ x,
                                              float* __restrict__ out) {
  __shared__ __attribute__((aligned(64))) _Float16 Wsh[2][64][32];
  const int lane = threadIdx.x & 31;
  const int wave = threadIdx.x >> 5;
  const int ln = lane & 15, lh = lane >> 4;
  const int t0 = blockIdx.x * 128 + wave * 16;
  const int obase = blockIdx.y * 64;

  const int st = threadIdx.x;
  const int srow = st >> 2, sg = st & 3, sgp = runswap(sg);
  const _Float16* wrow = wp + (size_t)(obase + srow) * 768 + sg * 8;
  const unsigned wdst[2] = { lds_off(&Wsh[0][srow][sgp * 8]),
                             lds_off(&Wsh[1][srow][sgp * 8]) };
  async_cp16(wdst[0], wrow);
  async_wait0();
  __syncthreads();

  v8f acc[4] = {};
  for (int kc = 0; kc < 24; ++kc) {
    const int buf = kc & 1;
    if (kc + 1 < 24)
      async_cp16(wdst[buf ^ 1], wrow + (kc + 1) * 32);
    if (kc + 2 < 24)
      __builtin_prefetch(wrow + (kc + 2) * 32, 0, 2);
    // xo is plain row-major -> two-run fragment load
    const v16h a = load_frag(xo + (size_t)(t0 + ln) * 768 + kc * 32 + (lh << 3));
    #pragma unroll
    for (int nt = 0; nt < 4; ++nt) {
      const v16h b = *(const v16h*)&Wsh[buf][nt * 16 + ln][lh << 4];
      acc[nt] = WMMA_F16(a, b, acc[nt]);
    }
    async_wait0();
    __syncthreads();
  }
  // (x + 0.3*(o-x)) / sqrt(0.58)
  #pragma unroll
  for (int j = 0; j < 4; ++j)
    #pragma unroll
    for (int vv = 0; vv < 8; ++vv) {
      const int tok = t0 + vv + 8 * lh;
      const int b = tok >> 8, p = tok & 255;
      const int oc = obase + 16 * j + ln;
      const size_t idx = ((size_t)b * 768 + oc) * 256 + p;
      out[idx] = (0.7f * x[idx] + 0.3f * acc[j][vv]) * 1.3130643285972254f;
    }
}

// ---------------------------------------------------------------------------
extern "C" void kernel_launch(void* const* d_in, const int* in_sizes, int n_in,
                              void* d_out, int out_size, void* d_ws, size_t ws_size,
                              hipStream_t stream) {
  (void)in_sizes; (void)n_in; (void)out_size; (void)ws_size;
  const float* x     = (const float*)d_in[0];
  const float* wqkv  = (const float*)d_in[1];
  const float* wproj = (const float*)d_in[2];
  // d_in[3] = batch_size (2) — shapes are compile-time constants here.

  char* ws = (char*)d_ws;
  _Float16* wq_h = (_Float16*)ws; ws += (size_t)2304 * 768 * 2;
  _Float16* wp_h = (_Float16*)ws; ws += (size_t)768 * 768 * 2;
  _Float16* xh   = (_Float16*)ws; ws += (size_t)8192 * 768 * 2;
  _Float16* qb   = (_Float16*)ws; ws += (size_t)24 * 4096 * 64 * 2;
  _Float16* kb   = (_Float16*)ws; ws += (size_t)24 * 4096 * 64 * 2;
  _Float16* vt   = (_Float16*)ws; ws += (size_t)24 * 4096 * 64 * 2;
  _Float16* xo   = (_Float16*)ws; ws += (size_t)8192 * 768 * 2;
  float* out = (float*)d_out;

  k_prep_w<<<288, 256, 0, stream>>>(wqkv, wq_h, 2304);
  k_prep_w<<<96,  256, 0, stream>>>(wproj, wp_h, 768);
  k_xpose<<<768, 256, 0, stream>>>(x, xh);
  k_qkv <<<dim3(64, 36), 256, 0, stream>>>(xh, wq_h, qb, kb, vt);
  k_attn<<<dim3(32, 24), 256, 0, stream>>>(qb, kb, vt, xo);
  k_proj<<<dim3(64, 12), 256, 0, stream>>>(xo, wp_h, x, out);
}
